// GraphTransformer_19344532701527
// MI455X (gfx1250) — matrix operand order repaired and checked by reading, not compiled, not run on hardware
//
#include <hip/hip_runtime.h>
#include <math.h>

typedef __attribute__((ext_vector_type(16))) __bf16 v16bf;
typedef __attribute__((ext_vector_type(8)))  float  v8f;

#define B_   2
#define L_   384
#define DIN  256
#define DE   128
#define H_   8
#define D_   32
#define HD   256
#define JT   64
#define NCH  (L_ / JT)
#define SCALE_ 0.17677669529663687f   // 32^-0.5

// ---------------- Kernel A: fused q,k,v,u projections ----------------
__global__ __launch_bounds__(256) void gt_proj(
    const float* __restrict__ x,   // (B,L,DIN)
    const float* __restrict__ Wq, const float* __restrict__ bq,
    const float* __restrict__ Wk, const float* __restrict__ bk,
    const float* __restrict__ Wv, const float* __restrict__ bv,
    const float* __restrict__ Wu, const float* __restrict__ bu,
    float* __restrict__ qo, float* __restrict__ ko,
    float* __restrict__ vo, float* __restrict__ uo)
{
    __shared__ float sX[DIN];
    const int row = blockIdx.x;      // 0..B*L-1
    const int c   = threadIdx.x;     // 0..255
    sX[c] = x[(size_t)row * DIN + c];
    __syncthreads();
    float aq = bq[c], ak = bk[c], av = bv[c], au = bu[c];
    #pragma unroll 4
    for (int f = 0; f < DIN; ++f) {
        const float  xv = sX[f];
        const size_t o  = (size_t)f * HD + c;
        aq = fmaf(xv, Wq[o], aq);
        ak = fmaf(xv, Wk[o], ak);
        av = fmaf(xv, Wv[o], av);
        au = fmaf(xv, Wu[o], au);
    }
    const size_t o = (size_t)row * HD + c;
    qo[o] = aq; ko[o] = ak; vo[o] = av; uo[o] = au;
}

// ------------- Kernel B: fused edge-GEMM + flash attention -------------
// One workgroup (8 waves / 256 threads) per (b, i) row.
__global__ __launch_bounds__(256) void gt_attn(
    const float* __restrict__ edge_feat,  // (B,L,L,DE)
    const float* __restrict__ edge_mask,  // (B,L,L)
    const float* __restrict__ We,         // (DE, HD)
    const float* __restrict__ qw, const float* __restrict__ kw,
    const float* __restrict__ vw, const float* __restrict__ uw,
    float* __restrict__ out)              // (B,L,HD)
{
    __shared__ __align__(16) __bf16 sA[JT][DE];   // 16 KB: edge chunk (bf16)
    __shared__ float sE[JT][HD];                  // 64 KB: e_i chunk (f32)
    __shared__ float sAtt[H_][JT];                // 2 KB : softmax probs
    __shared__ float sQ[HD];                      // q row

    const int tid  = threadIdx.x;
    const int wave = tid >> 5;
    const int lane = tid & 31;
    const int bi   = blockIdx.x;          // b*L + i
    const int b    = bi / L_;

    sQ[tid] = qw[(size_t)bi * HD + tid];

    // Preload We B-fragments in registers. Wave w owns e columns [32w,32w+32).
    // B layout (32x16 bf16): lane&15 = column, lanes<16 hold K=0..7,16..23; lanes>=16 hold K=8..15,24..31.
    v16bf bf[2][4];
    {
        const int kb2 = (lane & 16) ? 8 : 0;
        #pragma unroll
        for (int nt = 0; nt < 2; ++nt) {
            const int col = wave * 32 + nt * 16 + (lane & 15);
            #pragma unroll
            for (int ks = 0; ks < 4; ++ks) {
                v16bf f;
                #pragma unroll
                for (int t = 0; t < 16; ++t)
                    f[t] = (__bf16)We[(size_t)(ks * 32 + kb2 + t + (t >= 8 ? 8 : 0)) * HD + col];
                bf[nt][ks] = f;
            }
        }
    }

    const float* efb = edge_feat + (size_t)bi * L_ * DE;
    const float* mkb = edge_mask + (size_t)bi * L_;
    const int    h   = wave;              // wave <-> head
    const int    c   = tid;               // thread <-> output channel

    float m_state = -INFINITY;            // uniform per wave
    float l_state = 0.f;                  // uniform per wave
    float acc     = 0.f;                  // per-thread output accumulator

    for (int ch = 0; ch < NCH; ++ch) {
        const int j0 = ch * JT;
        __syncthreads();                  // previous chunk fully consumed

        // Stage edge chunk (64x128) as bf16, coalesced.
        for (int e = tid; e < JT * DE; e += 256)
            ((__bf16*)sA)[e] = (__bf16)efb[(size_t)j0 * DE + e];
        __syncthreads();

        // e_chunk(64x256) = A(64x128) @ We(128x256) via bf16 WMMA.
        #pragma unroll
        for (int mt = 0; mt < 4; ++mt) {
            v8f c0 = {0.f,0.f,0.f,0.f,0.f,0.f,0.f,0.f};
            v8f c1 = {0.f,0.f,0.f,0.f,0.f,0.f,0.f,0.f};
            const int row = mt * 16 + (lane & 15);
            const int kb  = (lane & 16) ? 8 : 0;  // A layout: lanes>=16 -> K+8
            #pragma unroll
            for (int ks = 0; ks < 4; ++ks) {
                union { v16bf v; float4 q[2]; } af;
                const __bf16* ap = &sA[row][ks * 32 + kb];
                af.q[0] = *reinterpret_cast<const float4*>(ap);       // K+0..7
                af.q[1] = *reinterpret_cast<const float4*>(ap + 16);  // K+16..23
                c0 = __builtin_amdgcn_wmma_f32_16x16x32_bf16(
                        false, af.v, false, bf[0][ks], (short)0, c0, false, false);
                c1 = __builtin_amdgcn_wmma_f32_16x16x32_bf16(
                        false, af.v, false, bf[1][ks], (short)0, c1, false, false);
            }
            // C/D layout: VGPR r, lanes 0-15 -> M=r, lanes 16-31 -> M=r+8.
            const int n0 = wave * 32 + (lane & 15);
            const int mr = mt * 16 + ((lane & 16) ? 8 : 0);
            #pragma unroll
            for (int r = 0; r < 8; ++r) {
                sE[mr + r][n0]      = c0[r];
                sE[mr + r][n0 + 16] = c1[r];
            }
        }
        __syncthreads();

        // Logits for head h = wave: logit[j] = q_h . (k_j + e_ij), mask+scale.
        float lg[2];
        #pragma unroll
        for (int half = 0; half < 2; ++half) {
            const int    jj = lane + half * 32;
            const float* kp = kw + ((size_t)b * L_ + (j0 + jj)) * HD + h * D_;
            const float* ep = &sE[jj][h * D_];
            const float* qp = &sQ[h * D_];
            float s = 0.f;
            #pragma unroll 8
            for (int d = 0; d < D_; ++d)
                s = fmaf(qp[d], kp[d] + ep[d], s);
            const float msk = mkb[j0 + jj];
            lg[half] = s * SCALE_ + (1.f - msk) * -1e9f;
        }
        // Online softmax (wave-wide, shuffle reductions).
        float wmax = fmaxf(lg[0], lg[1]);
        #pragma unroll
        for (int off = 16; off > 0; off >>= 1)
            wmax = fmaxf(wmax, __shfl_xor(wmax, off, 32));
        const float m_new = fmaxf(m_state, wmax);
        const float resc  = __expf(m_state - m_new);   // exp(-inf)=0 first chunk
        const float p0    = __expf(lg[0] - m_new);
        const float p1    = __expf(lg[1] - m_new);
        sAtt[h][lane]      = p0;
        sAtt[h][lane + 32] = p1;
        float ps = p0 + p1;
        #pragma unroll
        for (int off = 16; off > 0; off >>= 1)
            ps += __shfl_xor(ps, off, 32);
        l_state = l_state * resc + ps;
        m_state = m_new;
        __syncthreads();

        // acc[c] = acc[c]*resc + sum_j p[h][j] * (v[j][c] + e[j][c]); h = c>>5 = wave.
        acc *= resc;
        const float* vp = vw + ((size_t)b * L_ + j0) * HD + c;
        for (int j = 0; j < JT; ++j)
            acc = fmaf(sAtt[h][j], vp[(size_t)j * HD] + sE[j][c], acc);
    }

    out[(size_t)bi * HD + c] = acc / l_state + uw[(size_t)bi * HD + c];
}

extern "C" void kernel_launch(void* const* d_in, const int* in_sizes, int n_in,
                              void* d_out, int out_size, void* d_ws, size_t ws_size,
                              hipStream_t stream) {
    const float* node_feat = (const float*)d_in[0];
    const float* edge_feat = (const float*)d_in[1];
    const float* edge_mask = (const float*)d_in[2];
    const float* Wq = (const float*)d_in[3];
    const float* bq = (const float*)d_in[4];
    const float* Wk = (const float*)d_in[5];
    const float* bk = (const float*)d_in[6];
    const float* Wv = (const float*)d_in[7];
    const float* bv = (const float*)d_in[8];
    const float* We = (const float*)d_in[9];
    const float* Wu = (const float*)d_in[10];
    const float* bu = (const float*)d_in[11];
    float* out = (float*)d_out;

    float*       ws = (float*)d_ws;
    const size_t n  = (size_t)B_ * L_ * HD;   // 196608 floats per tensor
    float* qw = ws;
    float* kw = qw + n;
    float* vw = kw + n;
    float* uw = vw + n;

    gt_proj<<<B_ * L_, 256, 0, stream>>>(node_feat, Wq, bq, Wk, bk, Wv, bv,
                                         Wu, bu, qw, kw, vw, uw);
    gt_attn<<<B_ * L_, 256, 0, stream>>>(edge_feat, edge_mask, We,
                                         qw, kw, vw, uw, out);
}
